// CrossAttention_5257039970993
// MI455X (gfx1250) — compile-verified, matrix operand order checked by reading, not run on hardware
//
#include <hip/hip_runtime.h>
#include <stdint.h>

// ---------------- problem constants ----------------
#define BB 16
#define CC 512
#define NN 4096
#define HH 8
#define CN  (CC * NN)       // 2,097,152 elements per batch image
#define CSQ (CC * CC)       // 262,144 elements per CxC matrix

// ---------------- WMMA types ----------------
typedef __bf16 bf16_t;
typedef bf16_t v16bf __attribute__((ext_vector_type(16)));
typedef float  v8f   __attribute__((ext_vector_type(8)));
typedef float  v16f  __attribute__((ext_vector_type(16)));

#define WMMA_BF16(a, b, c) \
    __builtin_amdgcn_wmma_f32_16x16x32_bf16(false, (a), false, (b), (short)0, (c), false, false)

// ---------------- fragment loaders (CDNA5 16-bit A/B layout, wave32) ----------------
// A 16x32 bf16: lanes 0-15 hold row m, K {0..7} in v0-3 and {16..23} in v4-7;
// lanes 16-31 hold row m with K {8..15} and {24..31}.  Bt loads identically
// (lane j holds B^T row j = column j of B).
__device__ __forceinline__ v16bf ld_frag_bf(const bf16_t* p0, int ld) {
    int lane = threadIdx.x & 31;
    const bf16_t* p = p0 + (lane & 15) * ld + ((lane & 16) ? 8 : 0);
    union { uint4 q[2]; v16bf v; } u;
    u.q[0] = *(const uint4*)(p);
    u.q[1] = *(const uint4*)(p + 16);
    return u.v;
}

// same fragment from fp32 memory; hardware cvt (v_cvt_pk_bf16_f32) via convertvector
__device__ __forceinline__ v16bf ld_frag_f32cvt(const float* p0, int ld) {
    int lane = threadIdx.x & 31;
    const float* p = p0 + (lane & 15) * ld + ((lane & 16) ? 8 : 0);
    union { float4 f[4]; v16f vf; } u;
    u.f[0] = *(const float4*)(p);
    u.f[1] = *(const float4*)(p + 4);
    u.f[2] = *(const float4*)(p + 16);
    u.f[3] = *(const float4*)(p + 20);
    return __builtin_convertvector(u.vf, v16bf);
}

// ---------------- accumulator stores (16x16 f32 D layout) ----------------
// VGPR r: lanes 0-15 -> (M=r, N=lane); lanes 16-31 -> (M=8+r, N=lane-16)
__device__ __forceinline__ void st_acc_f32(float* C, int ld, v8f a) {
    int lane = threadIdx.x & 31;
    int n = lane & 15, mo = (lane & 16) ? 8 : 0;
#pragma unroll
    for (int r = 0; r < 8; ++r) C[(mo + r) * ld + n] = a[r];
}
__device__ __forceinline__ void st_acc_bf16(bf16_t* C, int ld, v8f a) {
    int lane = threadIdx.x & 31;
    int n = lane & 15, mo = (lane & 16) ? 8 : 0;
#pragma unroll
    for (int r = 0; r < 8; ++r) C[(mo + r) * ld + n] = (bf16_t)a[r];
}
// transposed tile store: C base points at element (n0, m0) of the transposed matrix
__device__ __forceinline__ void st_acc_bf16_T(bf16_t* C, int ld, v8f a) {
    int lane = threadIdx.x & 31;
    int n = lane & 15, mo = (lane & 16) ? 8 : 0;
#pragma unroll
    for (int r = 0; r < 8; ++r) C[n * ld + mo + r] = (bf16_t)a[r];
}

// ---------------- generic per-wave 64x64 bf16 GEMM (4x4 WMMA subtiles) ----------------
__device__ __forceinline__ void wave_gemm_bf16(const bf16_t* A, int lda,
                                               const bf16_t* BT, int ldb,
                                               int K, v8f* acc) {
    for (int k = 0; k < K; k += 32) {
        v16bf a[4], b[4];
#pragma unroll
        for (int i = 0; i < 4; ++i) a[i] = ld_frag_bf(A + i * 16 * lda + k, lda);
#pragma unroll
        for (int j = 0; j < 4; ++j) b[j] = ld_frag_bf(BT + j * 16 * ldb + k, ldb);
#pragma unroll
        for (int i = 0; i < 4; ++i)
#pragma unroll
            for (int j = 0; j < 4; ++j)
                acc[i * 4 + j] = WMMA_BF16(a[i], b[j], acc[i * 4 + j]);
    }
}

__device__ __forceinline__ void zero_acc(v8f* acc) {
    v8f z = {0.f, 0.f, 0.f, 0.f, 0.f, 0.f, 0.f, 0.f};
#pragma unroll
    for (int t = 0; t < 16; ++t) acc[t] = z;
}

// ============================================================================
// K0: convert the 8 needed weight matrices to bf16 (Wv's stored transposed).
// slot: 0=wq_rgb 1=wk_rgb 2=wvT_rgb 3=wq_point 4=wk_point 5=wvT_point
//       6=proj_r2p 7=proj_p2r
// ============================================================================
__global__ void k_prep(const float* wq_rgb, const float* wk_rgb, const float* wv_rgb,
                       const float* wq_point, const float* wk_point, const float* wv_point,
                       const float* proj_r2p, const float* proj_p2r, bf16_t* wbf) {
    int idx = blockIdx.x * 256 + threadIdx.x;      // 8 * 262144 total
    int mat = idx >> 18, pos = idx & (CSQ - 1);
    int r = pos >> 9, c = pos & 511;
    float v;
    switch (mat) {
        case 0: v = wq_rgb[pos]; break;
        case 1: v = wk_rgb[pos]; break;
        case 2: v = wv_rgb[c * CC + r]; break;     // transpose
        case 3: v = wq_point[pos]; break;
        case 4: v = wk_point[pos]; break;
        case 5: v = wv_point[c * CC + r]; break;   // transpose
        case 6: v = proj_r2p[pos]; break;
        default: v = proj_p2r[pos]; break;
    }
    wbf[idx] = (bf16_t)v;
}

// ============================================================================
// Gram: G[b][c1][c2] = sum_n rgbE[b][c1][n] * geoE[b][c2][n]   (K = 4096)
// Both operands contiguous along n -> no transposes.  Also writes GT.
// ============================================================================
__global__ void __launch_bounds__(256, 1)
k_gram(const float* rgbE, const float* geoE, bf16_t* G, bf16_t* GT) {
    int b = blockIdx.z, w = threadIdx.x >> 5;
    int m0 = blockIdx.y * 128 + (w & 1) * 64;
    int n0 = blockIdx.x * 256 + (w >> 1) * 64;
    const float* A  = rgbE + (size_t)b * CN + (size_t)m0 * NN;
    const float* Bt = geoE + (size_t)b * CN + (size_t)n0 * NN;
    v8f acc[16]; zero_acc(acc);
    for (int k = 0; k < NN; k += 32) {
        v16bf a[4], bf[4];
#pragma unroll
        for (int i = 0; i < 4; ++i) a[i] = ld_frag_f32cvt(A + i * 16 * NN + k, NN);
#pragma unroll
        for (int j = 0; j < 4; ++j) bf[j] = ld_frag_f32cvt(Bt + j * 16 * NN + k, NN);
#pragma unroll
        for (int i = 0; i < 4; ++i)
#pragma unroll
            for (int j = 0; j < 4; ++j)
                acc[i * 4 + j] = WMMA_BF16(a[i], bf[j], acc[i * 4 + j]);
    }
    bf16_t* g  = G  + (size_t)b * CSQ;
    bf16_t* gt = GT + (size_t)b * CSQ;
#pragma unroll
    for (int i = 0; i < 4; ++i)
#pragma unroll
        for (int j = 0; j < 4; ++j) {
            st_acc_bf16  (g  + (m0 + i * 16) * CC + n0 + j * 16, CC, acc[i * 4 + j]);
            st_acc_bf16_T(gt + (n0 + j * 16) * CC + m0 + i * 16, CC, acc[i * 4 + j]);
        }
}

// ============================================================================
// C1: T[z] = Wq * (G or GT)    z = b*2+br,  br0: Wq_rgb*G, br1: Wq_point*G^T
// ============================================================================
__global__ void __launch_bounds__(256, 1)
k_c1(const bf16_t* wbf, const bf16_t* G, const bf16_t* GT, bf16_t* T) {
    int z = blockIdx.z, b = z >> 1, br = z & 1, w = threadIdx.x >> 5;
    int m0 = blockIdx.y * 128 + (w & 1) * 64;
    int n0 = blockIdx.x * 256 + (w >> 1) * 64;
    const bf16_t* A  = wbf + (size_t)(br ? 3 : 0) * CSQ + m0 * CC;
    const bf16_t* Bt = (br ? G : GT) + (size_t)b * CSQ + n0 * CC;
    v8f acc[16]; zero_acc(acc);
    wave_gemm_bf16(A, CC, Bt, CC, CC, acc);
    bf16_t* out = T + (size_t)z * CSQ;
#pragma unroll
    for (int i = 0; i < 4; ++i)
#pragma unroll
        for (int j = 0; j < 4; ++j)
            st_acc_bf16(out + (m0 + i * 16) * CC + n0 + j * 16, CC, acc[i * 4 + j]);
}

// ============================================================================
// C2: per (b,br,h) 64x64 diagonal score block  S_h = T_h-rows * Wk_h-rows^T
// ============================================================================
__global__ void __launch_bounds__(256, 1)
k_c2(const bf16_t* T, const bf16_t* wbf, float* S) {
    int w = blockIdx.x * 8 + (threadIdx.x >> 5);     // 0..255
    int b = w >> 4, br = (w >> 3) & 1, h = w & 7, z = b * 2 + br;
    const bf16_t* A  = T + (size_t)z * CSQ + h * 64 * CC;
    const bf16_t* Bt = wbf + (size_t)(br ? 1 : 4) * CSQ + h * 64 * CC;
    v8f acc[16]; zero_acc(acc);
    wave_gemm_bf16(A, CC, Bt, CC, CC, acc);
    float* out = S + (size_t)(z * 8 + h) * 4096;
#pragma unroll
    for (int i = 0; i < 4; ++i)
#pragma unroll
        for (int j = 0; j < 4; ++j)
            st_acc_f32(out + (i * 16) * 64 + j * 16, 64, acc[i * 4 + j]);
}

// ============================================================================
// C3: softmax over last dim (64) with scale d^-0.5 = 0.125; one wave per row
// ============================================================================
__global__ void k_c3(const float* S, bf16_t* attn) {
    int row = blockIdx.x * 8 + (threadIdx.x >> 5);
    int lane = threadIdx.x & 31;
    const float* r = S + (size_t)row * 64;
    float x0 = r[lane] * 0.125f, x1 = r[lane + 32] * 0.125f;
    float mx = fmaxf(x0, x1);
#pragma unroll
    for (int o = 16; o; o >>= 1) mx = fmaxf(mx, __shfl_xor(mx, o, 32));
    float e0 = __expf(x0 - mx), e1 = __expf(x1 - mx);
    float s = e0 + e1;
#pragma unroll
    for (int o = 16; o; o >>= 1) s += __shfl_xor(s, o, 32);
    float inv = 1.f / s;
    attn[(size_t)row * 64 + lane]      = (bf16_t)(e0 * inv);
    attn[(size_t)row * 64 + lane + 32] = (bf16_t)(e1 * inv);
}

// ============================================================================
// C4: M[z][cout][h*64+dd] = sum_e proj[cout][h*64+e] * attn_h[dd][e]   (K=64)
// ============================================================================
__global__ void __launch_bounds__(256, 1)
k_c4(const bf16_t* wbf, const bf16_t* attn, bf16_t* M) {
    int w = blockIdx.x * 8 + (threadIdx.x >> 5);     // 0..2047
    int b = w >> 7, br = (w >> 6) & 1, h = (w >> 3) & 7, cb = w & 7;
    int z = b * 2 + br;
    const bf16_t* A  = wbf + (size_t)(br ? 6 : 7) * CSQ + (cb * 64) * CC + h * 64;
    const bf16_t* Bt = attn + (size_t)(z * 8 + h) * 4096;
    v8f acc[16]; zero_acc(acc);
    wave_gemm_bf16(A, CC, Bt, 64, 64, acc);
    bf16_t* out = M + (size_t)z * CSQ;
#pragma unroll
    for (int i = 0; i < 4; ++i)
#pragma unroll
        for (int j = 0; j < 4; ++j)
            st_acc_bf16(out + (cb * 64 + i * 16) * CC + h * 64 + j * 16, CC, acc[i * 4 + j]);
}

// ============================================================================
// C5: R = M * Wv  (+ identity), packed into Rc[b][C][2C]:
//     br1 (r2p, v=rgb) -> cols 0..511 (multiplies rgbE); br0 -> cols 512..1023
// ============================================================================
__global__ void __launch_bounds__(256, 1)
k_c5(const bf16_t* M, const bf16_t* wbf, bf16_t* Rc) {
    int z = blockIdx.z, b = z >> 1, br = z & 1, w = threadIdx.x >> 5;
    int m0 = blockIdx.y * 128 + (w & 1) * 64;
    int n0 = blockIdx.x * 256 + (w >> 1) * 64;
    const bf16_t* A  = M + (size_t)z * CSQ + m0 * CC;
    const bf16_t* Bt = wbf + (size_t)(br ? 2 : 5) * CSQ + n0 * CC;
    v8f acc[16]; zero_acc(acc);
    wave_gemm_bf16(A, CC, Bt, CC, CC, acc);
    int colbase = br ? 0 : CC;
    bf16_t* out = Rc + (size_t)b * (2 * CSQ);
    int lane = threadIdx.x & 31;
    int nl = lane & 15, mo = (lane & 16) ? 8 : 0;
#pragma unroll
    for (int i = 0; i < 4; ++i)
#pragma unroll
        for (int j = 0; j < 4; ++j)
#pragma unroll
            for (int r = 0; r < 8; ++r) {
                int row = m0 + i * 16 + mo + r;
                int col = n0 + j * 16 + nl;
                float v = acc[i * 4 + j][r] + ((row == col) ? 1.f : 0.f);
                out[(size_t)row * 1024 + colbase + col] = (bf16_t)v;
            }
}

// ============================================================================
// Final: out[b] = Rc[b] (C x 2C) * [rgbE[b]; geoE[b]] (2C x N), f32 out.
// B operand staged through double-buffered LDS (coalesced f32 reads -> packed
// bf16 pairs, transposed).  Block tile 128c x 256n, 8 waves of 64x64,
// K=1024 in chunks of 32; fill of chunk k+1 overlaps WMMAs of chunk k.
// ============================================================================
#define BT_PITCH 40   // halfwords; multiple of 8 keeps 16B alignment, spreads banks
__global__ void __launch_bounds__(256, 1)
k_final(const float* rgbE, const float* geoE, const bf16_t* Rc, float* out) {
    __shared__ bf16_t Bt[2][256 * BT_PITCH];
    int b = blockIdx.z;
    int cblk = blockIdx.y * 128;
    int nblk = blockIdx.x * 256;
    int w = threadIdx.x >> 5, wm = w & 1, wn = w >> 1;
    const bf16_t* Ab = Rc + (size_t)b * (2 * CSQ) + (cblk + wm * 64) * 1024;
    const float* rgbB = rgbE + (size_t)b * CN + nblk;
    const float* geoB = geoE + (size_t)b * CN + nblk;
    int n = threadIdx.x;                           // each thread owns one LDS row

    auto fill = [&](int kc, int buf) {
        int k0 = kc * 32;
        const float* src = (k0 < CC) ? rgbB + (size_t)k0 * NN
                                     : geoB + (size_t)(k0 - CC) * NN;
        uint32_t* Brow = (uint32_t*)&Bt[buf][n * BT_PITCH];
#pragma unroll 4
        for (int kk2 = 0; kk2 < 16; ++kk2) {
            union { bf16_t h[2]; uint32_t u; } pk;
            pk.h[0] = (bf16_t)src[(size_t)(2 * kk2) * NN + n];
            pk.h[1] = (bf16_t)src[(size_t)(2 * kk2 + 1) * NN + n];
            Brow[kk2] = pk.u;
        }
    };

    v8f acc[16]; zero_acc(acc);
    fill(0, 0);
    const int NCH = 1024 / 32;
    for (int kc = 0; kc < NCH; ++kc) {
        __syncthreads();                           // buf[kc&1] ready; prev reads done
        if (kc + 1 < NCH) fill(kc + 1, (kc + 1) & 1);
        int k0 = kc * 32, cur = kc & 1;
        v16bf a[4], bfr[4];
#pragma unroll
        for (int i = 0; i < 4; ++i) a[i] = ld_frag_bf(Ab + i * 16 * 1024 + k0, 1024);
#pragma unroll
        for (int j = 0; j < 4; ++j)
            bfr[j] = ld_frag_bf(&Bt[cur][(wn * 64 + j * 16) * BT_PITCH], BT_PITCH);
#pragma unroll
        for (int i = 0; i < 4; ++i)
#pragma unroll
            for (int j = 0; j < 4; ++j)
                acc[i * 4 + j] = WMMA_BF16(a[i], bfr[j], acc[i * 4 + j]);
    }
    float* o = out + (size_t)b * CN;
#pragma unroll
    for (int i = 0; i < 4; ++i)
#pragma unroll
        for (int j = 0; j < 4; ++j)
            st_acc_f32(o + (size_t)(cblk + wm * 64 + i * 16) * NN + nblk + wn * 64 + j * 16,
                       NN, acc[i * 4 + j]);
}

// ============================================================================
// launcher
// ============================================================================
extern "C" void kernel_launch(void* const* d_in, const int* in_sizes, int n_in,
                              void* d_out, int out_size, void* d_ws, size_t ws_size,
                              hipStream_t stream) {
    const float* rgbE     = (const float*)d_in[0];
    const float* geoE     = (const float*)d_in[1];
    const float* wq_rgb   = (const float*)d_in[2];
    const float* wk_rgb   = (const float*)d_in[3];
    const float* wv_rgb   = (const float*)d_in[4];
    const float* wq_point = (const float*)d_in[5];
    const float* wk_point = (const float*)d_in[6];
    const float* wv_point = (const float*)d_in[7];
    const float* proj_r2p = (const float*)d_in[8];
    const float* proj_p2r = (const float*)d_in[9];
    float* outp = (float*)d_out;

    // workspace layout (lifetime-aliased, ~42 MB total)
    char* ws = (char*)d_ws;
    const size_t OFF_GM   = 4u << 20;                 // after 4MB of bf16 weights
    const size_t OFF_TR   = OFF_GM + (16u << 20);
    const size_t OFF_S    = OFF_TR + (16u << 20);
    const size_t OFF_ATTN = OFF_S + (4u << 20);
    bf16_t* wbf  = (bf16_t*)ws;
    bf16_t* G    = (bf16_t*)(ws + OFF_GM);
    bf16_t* GT   = G + (size_t)BB * CSQ;
    bf16_t* Mmat = G;                  // aliases G/GT (dead after k_c1)
    bf16_t* T    = (bf16_t*)(ws + OFF_TR);
    bf16_t* Rc   = T;                  // aliases T (dead after k_c2)
    float*  S    = (float*)(ws + OFF_S);
    bf16_t* attn = (bf16_t*)(ws + OFF_ATTN);

    k_prep<<<dim3(8 * CSQ / 256), dim3(256), 0, stream>>>(
        wq_rgb, wk_rgb, wv_rgb, wq_point, wk_point, wv_point, proj_r2p, proj_p2r, wbf);
    k_gram<<<dim3(2, 4, BB), dim3(256), 0, stream>>>(rgbE, geoE, G, GT);
    k_c1  <<<dim3(2, 4, 2 * BB), dim3(256), 0, stream>>>(wbf, G, GT, T);
    k_c2  <<<dim3(32), dim3(256), 0, stream>>>(T, wbf, S);
    k_c3  <<<dim3(2048), dim3(256), 0, stream>>>(S, attn);
    k_c4  <<<dim3(256), dim3(256), 0, stream>>>(wbf, attn, Mmat);
    k_c5  <<<dim3(2, 4, 2 * BB), dim3(256), 0, stream>>>(Mmat, wbf, Rc);
    k_final<<<dim3(16, 4, BB), dim3(256), 0, stream>>>(rgbE, geoE, Rc, outp);
}
